// DeepConvAE_29661044146821
// MI455X (gfx1250) — compile-verified, matrix operand order checked by reading, not run on hardware
//
#include <hip/hip_runtime.h>
#include <hip/hip_bf16.h>
#include <math.h>

// ---------------------------------------------------------------------------
// DeepConvAE on gfx1250 (MI455X, wave32).
// All channel contractions via v_wmma_f32_16x16x32_f16 (f32 accumulate).
// Activations: f16 NHWC with zeroed 4-px halo + 16-aligned right pad (Wp=72)
//   => inner-loop loads need NO bounds checks; EXEC all-ones around WMMAs.
// 128-channel layers: block = 4 waves = one 32-px M-supertile x all 128 ch.
//   5x36xC input patch staged once in LDS (46KB, ds_load_b128 A-fragments).
//   Each wave register-tiles 2x2 (32px x 32ch): 4 independent WMMAs per
//   (ky,kx,kt) step sharing 4 LDS + 2 global loads => 32B/WMMA.
// Weights: device-prepacked to per-lane B-fragment layout (aligned v16h).
// ---------------------------------------------------------------------------

typedef __attribute__((ext_vector_type(16))) _Float16 v16h;
typedef __attribute__((ext_vector_type(8)))  _Float16 v8h;
typedef __attribute__((ext_vector_type(8)))  float    v8f;

#define WP 72   // padded row width (halo 4 + image + tile pad + 4)

#define WMMA_F16(A, B, C) __builtin_amdgcn_wmma_f32_16x16x32_f16( \
        false, (A), false, (B), (short)0, (C), false, false)

// ---------------------------------------------------------------------------
// LDS-staged, 2x2 register-tiled WMMA conv. OCpad=128, bias+ReLU, f16 NHWC.
// KT = ICpad/32 (1 or 4), PAD = 0 (conv) or 4 (transposed conv).
// Block: 128 thr = 4 waves; block -> (b, oy, 32px supertile); wave w -> 32ch.
// ---------------------------------------------------------------------------
template<int KT, int PAD>
__global__ void __launch_bounds__(128)
wmma_conv_tile(const _Float16* __restrict__ in, const _Float16* __restrict__ wpk,
               const float* __restrict__ bias, _Float16* __restrict__ outh,
               int Hp_in, int OH, int OW, int Hp_out)
{
    constexpr int ICp = KT * 32;
    __shared__ _Float16 smem[5 * 36 * ICp];      // 5 rows x 36 px x ICp ch

    int t = blockIdx.x;
    const int ms = t & 1; t >>= 1;               // M supertile: px base ms*32
    const int oy = t % OH;
    const int b  = t / OH;
    const int ox0 = ms << 5;

    // ---- cooperative stage: rows oy-PAD..+4, px ox0-PAD..+35 (halo-safe) ----
    {
        const int cpc = ICp / 8;                 // 16B chunks per pixel
        const int chunks = 5 * 36 * cpc;
        const size_t rowbase =
            ((size_t)(b * Hp_in + (oy - PAD + 4)) * WP + (ox0 - PAD + 4)) * ICp;
        for (int i = threadIdx.x; i < chunks; i += 128) {
            const int cc  = i % cpc;
            const int pix = i / cpc;             // r*36 + px
            const int px  = pix % 36, r = pix / 36;
            const v8h* src = (const v8h*)(in + rowbase +
                              ((size_t)r * WP + px) * ICp) + cc;
            *((v8h*)(smem + (size_t)pix * ICp) + cc) = *src;
        }
    }
    __syncthreads();

    const int lane = threadIdx.x & 31;
    const int ns   = threadIdx.x >> 5;           // N supertile (32 channels)
    const int mrow = lane & 15;                  // A: lane's M row
    const int khf  = lane >> 4;                  // A: K-half select
    const int nt0  = ns * 2, nt1 = nt0 + 1;

    v8f acc00 = {}, acc01 = {}, acc10 = {}, acc11 = {};

    for (int ky = 0; ky < 5; ++ky) {
        #pragma unroll
        for (int kx = 0; kx < 5; ++kx) {
            const _Float16* arow0 = smem + (size_t)(ky * 36 + mrow + kx) * ICp;
            const _Float16* arow1 = arow0 + (size_t)16 * ICp;
            #pragma unroll
            for (int kt = 0; kt < KT; ++kt) {
                v8h lo0 = *(const v8h*)(arow0 + kt * 32 + khf * 8);
                v8h hi0 = *(const v8h*)(arow0 + kt * 32 + 16 + khf * 8);
                v8h lo1 = *(const v8h*)(arow1 + kt * 32 + khf * 8);
                v8h hi1 = *(const v8h*)(arow1 + kt * 32 + 16 + khf * 8);
                v16h a0, a1;
                #pragma unroll
                for (int i = 0; i < 8; ++i) {
                    a0[i] = lo0[i]; a0[8 + i] = hi0[i];
                    a1[i] = lo1[i]; a1[8 + i] = hi1[i];
                }
                const _Float16* wb = wpk +
                    (((size_t)(ky * 5 + kx) * KT + kt) * 8) * 512;
                v16h b0 = *(const v16h*)(wb + ((size_t)nt0 * 32 + lane) * 16);
                v16h b1 = *(const v16h*)(wb + ((size_t)nt1 * 32 + lane) * 16);
                acc00 = WMMA_F16(a0, b0, acc00);
                acc01 = WMMA_F16(a0, b1, acc01);
                acc10 = WMMA_F16(a1, b0, acc10);
                acc11 = WMMA_F16(a1, b1, acc11);
            }
        }
    }

    // C/D layout: lane L -> N = L&15 ; VGPR v -> M = (L>>4)*8 + v
    const size_t obase = ((size_t)(b * Hp_out + oy + 4) * WP + 4) * 128;
    auto store_tile = [&](const v8f& acc, int nt, int mbase) {
        const int oc = (nt << 4) + (lane & 15);
        const float bv = bias[oc];
        #pragma unroll
        for (int v = 0; v < 8; ++v) {
            const int oxv = mbase + khf * 8 + v;
            if (oxv < OW) {
                float r = acc[v] + bv;
                r = r > 0.0f ? r : 0.0f;
                outh[obase + (size_t)oxv * 128 + oc] = (_Float16)r;
            }
        }
    };
    store_tile(acc00, nt0, ox0);
    store_tile(acc01, nt1, ox0);
    store_tile(acc10, nt0, ox0 + 16);
    store_tile(acc11, nt1, ox0 + 16);
}

// ---------------------------------------------------------------------------
// Final transposed conv: 60x60x128 -> 64x64x3, sigmoid, f32 NCHW to d_out.
// Direct global A loads (halo => no bounds checks). KT=4, PAD=4, Nt=1.
// ---------------------------------------------------------------------------
__global__ void __launch_bounds__(256)
wmma_deconv_final(const _Float16* __restrict__ in,
                  const _Float16* __restrict__ wpk,
                  const float* __restrict__ bias, float* __restrict__ out)
{
    const int lane = threadIdx.x & 31;
    const int wid  = blockIdx.x * 8 + (threadIdx.x >> 5);   // 64*64*4 waves
    const int oxt = wid & 3;
    const int oy  = (wid >> 2) & 63;
    const int b   = wid >> 8;

    const int mrow = lane & 15;
    const int khf  = lane >> 4;
    const int ox   = (oxt << 4) + mrow;

    v8f acc = {};
    for (int ky = 0; ky < 5; ++ky) {
        const size_t ybase = (size_t)(b * 68 + (oy + ky)) * WP;   // y = oy+ky-4+4
        #pragma unroll
        for (int kx = 0; kx < 5; ++kx) {
            const _Float16* abase = in + (ybase + (ox + kx)) * 128;
            #pragma unroll
            for (int kt = 0; kt < 4; ++kt) {
                v8h lo = *(const v8h*)(abase + kt * 32 + khf * 8);
                v8h hi = *(const v8h*)(abase + kt * 32 + 16 + khf * 8);
                v16h a;
                #pragma unroll
                for (int i = 0; i < 8; ++i) { a[i] = lo[i]; a[8 + i] = hi[i]; }
                v16h bf = *(const v16h*)(wpk +
                    (((size_t)(ky * 5 + kx) * 4 + kt) * 32 + lane) * 16);
                acc = WMMA_F16(a, bf, acc);
            }
        }
    }

    const int oc = lane & 15;
    if (oc < 3) {
        const float bv = bias[oc];
        #pragma unroll
        for (int v = 0; v < 8; ++v) {
            const int oxv = (oxt << 4) + khf * 8 + v;
            float r = acc[v] + bv;
            r = 1.0f / (1.0f + expf(-r));
            out[((size_t)(b * 3 + oc) * 64 + oy) * 64 + oxv] = r;
        }
    }
}

// ---------------------------------------------------------------------------
// Pack f32 weights into per-lane B-fragment order (f16).
// dst[((((ky*5+kx)*Kt + kt)*Nt + nt)*32 + lane)*16 + h]
// lane: n = lane&15, khalf = lane>>4 ; h<8: k = khalf*8+h else 16+khalf*8+h-8
// flip=0 (conv, src OCxICx5x5): w[n][k][ky][kx]
// flip=1 (deconv, src ICxOCx5x5): w[k][n][4-ky][4-kx]
// ---------------------------------------------------------------------------
__global__ void pack_weights(const float* __restrict__ src,
                             _Float16* __restrict__ dst,
                             int IC, int OC, int ICpad, int OCpad, int flip)
{
    const int Kt = ICpad >> 5, Nt = OCpad >> 4;
    const size_t total = (size_t)25 * Kt * Nt * 512;
    const size_t i = (size_t)blockIdx.x * blockDim.x + threadIdx.x;
    if (i >= total) return;
    const int h    = (int)(i & 15);
    const int lane = (int)((i >> 4) & 31);
    size_t r = i >> 9;
    const int nt = (int)(r % Nt); r /= Nt;
    const int kt = (int)(r % Kt); r /= Kt;
    const int ky = (int)r / 5, kx = (int)r % 5;
    const int khalf = lane >> 4;
    const int klocal = (h < 8) ? (khalf * 8 + h) : (16 + khalf * 8 + (h - 8));
    const int K = kt * 32 + klocal;
    const int N = nt * 16 + (lane & 15);
    float v = 0.0f;
    if (K < IC && N < OC) {
        if (!flip) v = src[(((size_t)N * IC + K) * 5 + ky) * 5 + kx];
        else       v = src[(((size_t)K * OC + N) * 5 + (4 - ky)) * 5 + (4 - kx)];
    }
    dst[i] = (_Float16)v;
}

// x (f32 NCHW 64x3x64x64) -> a0 (f16 NHWC halo layout, C=32; zeros elsewhere
// provided by memset). Writes only the 3 real channels of interior pixels.
__global__ void convert_input(const float* __restrict__ x,
                              _Float16* __restrict__ a0)
{
    const int i = blockIdx.x * blockDim.x + threadIdx.x;   // 64*64*64 pixels
    if (i >= 64 * 64 * 64) return;
    const int xc = i & 63;
    const int y  = (i >> 6) & 63;
    const int b  = i >> 12;
    _Float16* dst = a0 + ((size_t)(b * 72 + y + 4) * WP + (xc + 4)) * 32;
    const float* sp = x + (size_t)b * 3 * 4096 + y * 64 + xc;
    dst[0] = (_Float16)sp[0];
    dst[1] = (_Float16)sp[4096];
    dst[2] = (_Float16)sp[8192];
}

// Per-(b,c) spatial max over the 52x52 interior (halo layout, Hp=60).
__global__ void channel_max(const _Float16* __restrict__ h,
                            float* __restrict__ m)
{
    const int b = blockIdx.x;
    const int c = threadIdx.x;                   // 0..127
    const _Float16* base = h + ((size_t)(b * 60 + 4) * WP + 4) * 128 + c;
    float mx = -1e30f;
    for (int y = 0; y < 52; ++y)
        for (int x = 0; x < 52; ++x) {
            const float v = (float)base[((size_t)y * WP + x) * 128];
            mx = v > mx ? v : mx;
        }
    m[b * 128 + c] = mx;
}

// Spatial-argmax mask + 4x4-block cross-channel max mask, in place.
// grid = 64*13*13 (one block per (b,by,bx)), block = 256.
__global__ void __launch_bounds__(256)
block_sparsify(_Float16* __restrict__ h, const float* __restrict__ m)
{
    __shared__ float red[256];
    const int blk = blockIdx.x;
    const int bx = blk % 13;
    const int t2 = blk / 13;
    const int by = t2 % 13;
    const int b  = t2 / 13;
    const int tid = threadIdx.x;

    float vals[8];
    float lmax = 0.0f;                           // masked values >= 0 (ReLU)
    #pragma unroll
    for (int j = 0; j < 8; ++j) {
        const int e = tid + j * 256;             // 16 positions x 128 channels
        const int c = e & 127;
        const int p = e >> 7;
        const int py = p >> 2, px = p & 3;
        const size_t idx = ((size_t)(b * 60 + by * 4 + py + 4) * WP +
                            (bx * 4 + px + 4)) * 128 + c;
        const float v  = (float)h[idx];
        const float mv = (v == m[b * 128 + c]) ? v : 0.0f;   // spatial sparsity
        vals[j] = mv;
        lmax = mv > lmax ? mv : lmax;
    }
    red[tid] = lmax;
    __syncthreads();
    for (int s = 128; s > 0; s >>= 1) {
        if (tid < s) { const float o = red[tid + s]; if (o > red[tid]) red[tid] = o; }
        __syncthreads();
    }
    const float mb = red[0];
    #pragma unroll
    for (int j = 0; j < 8; ++j) {
        const int e = tid + j * 256;
        const int c = e & 127;
        const int p = e >> 7;
        const int py = p >> 2, px = p & 3;
        const size_t idx = ((size_t)(b * 60 + by * 4 + py + 4) * WP +
                            (bx * 4 + px + 4)) * 128 + c;
        const float mv = vals[j];
        h[idx] = (_Float16)((mv == mb) ? mv : 0.0f);         // channel sparsity
    }
}

// ---------------------------------------------------------------------------
extern "C" void kernel_launch(void* const* d_in, const int* in_sizes, int n_in,
                              void* d_out, int out_size, void* d_ws, size_t ws_size,
                              hipStream_t stream)
{
    const float* x   = (const float*)d_in[0];
    const float* ew0 = (const float*)d_in[1];
    const float* eb0 = (const float*)d_in[2];
    const float* ew1 = (const float*)d_in[3];
    const float* eb1 = (const float*)d_in[4];
    const float* ew2 = (const float*)d_in[5];
    const float* eb2 = (const float*)d_in[6];
    const float* dw0 = (const float*)d_in[7];
    const float* db0 = (const float*)d_in[8];
    const float* dw1 = (const float*)d_in[9];
    const float* db1 = (const float*)d_in[10];
    const float* dw2 = (const float*)d_in[11];
    const float* db2 = (const float*)d_in[12];
    float* out = (float*)d_out;

    // Ping-pong halo buffers, each sized for the largest layer (Hp=68):
    // 64 * 68 * 72 * 128 * 2B = 80,216,064 B. Total ws ~164 MB.
    char* ws = (char*)d_ws;
    const size_t MAXB = (size_t)64 * 68 * WP * 128 * sizeof(_Float16);
    _Float16* X = (_Float16*)ws;
    _Float16* Y = (_Float16*)(ws + MAXB);
    char* p = ws + 2 * MAXB;
    _Float16* wp_e0 = (_Float16*)p; p += (size_t)25 * 1 * 8 * 512 * 2;
    _Float16* wp_e1 = (_Float16*)p; p += (size_t)25 * 4 * 8 * 512 * 2;
    _Float16* wp_e2 = (_Float16*)p; p += (size_t)25 * 4 * 8 * 512 * 2;
    _Float16* wp_d0 = (_Float16*)p; p += (size_t)25 * 4 * 8 * 512 * 2;
    _Float16* wp_d1 = (_Float16*)p; p += (size_t)25 * 4 * 8 * 512 * 2;
    _Float16* wp_d2 = (_Float16*)p; p += (size_t)25 * 4 * 1 * 512 * 2;
    float* mbuf = (float*)p;                                 // 64*128 f32

    auto packL = [&](const float* src, _Float16* dst, int IC, int OC,
                     int ICpad, int OCpad, int flip) {
        const size_t tot = (size_t)25 * (ICpad / 32) * (OCpad / 16) * 512;
        pack_weights<<<(unsigned)((tot + 255) / 256), 256, 0, stream>>>(
            src, dst, IC, OC, ICpad, OCpad, flip);
    };
    packL(ew0, wp_e0,   3, 128,  32, 128, 0);
    packL(ew1, wp_e1, 128, 128, 128, 128, 0);
    packL(ew2, wp_e2, 128, 128, 128, 128, 0);
    packL(dw0, wp_d0, 128, 128, 128, 128, 1);
    packL(dw1, wp_d1, 128, 128, 128, 128, 1);
    packL(dw2, wp_d2, 128,   3, 128,  16, 1);

    auto bufBytes = [](int Hp, int C) {
        return (size_t)64 * Hp * WP * C * sizeof(_Float16);
    };

    // a0: 64x64 image, C=32, Hp=72
    hipMemsetAsync(X, 0, bufBytes(72, 32), stream);
    convert_input<<<(64 * 64 * 64 + 255) / 256, 256, 0, stream>>>(x, X);

    auto convL = [&](auto kern, const _Float16* in, const _Float16* wpk,
                     const float* bias, _Float16* oh,
                     int Hp_in, int OH, int OW, int Hp_out) {
        const unsigned blocks = (unsigned)(64 * OH * 2);   // 2 x 32px supertiles
        kern<<<blocks, 128, 0, stream>>>(in, wpk, bias, oh,
                                         Hp_in, OH, OW, Hp_out);
    };

    // encoder
    hipMemsetAsync(Y, 0, bufBytes(68, 128), stream);                 // h0 60x60
    convL(wmma_conv_tile<1, 0>, X, wp_e0, eb0, Y, 72, 60, 60, 68);
    hipMemsetAsync(X, 0, bufBytes(64, 128), stream);                 // h1 56x56
    convL(wmma_conv_tile<4, 0>, Y, wp_e1, eb1, X, 68, 56, 56, 64);
    hipMemsetAsync(Y, 0, bufBytes(60, 128), stream);                 // h2 52x52
    convL(wmma_conv_tile<4, 0>, X, wp_e2, eb2, Y, 64, 52, 52, 60);

    // sparsify (in place on Y)
    channel_max<<<64, 128, 0, stream>>>(Y, mbuf);
    block_sparsify<<<64 * 13 * 13, 256, 0, stream>>>(Y, mbuf);

    // decoder
    hipMemsetAsync(X, 0, bufBytes(64, 128), stream);                 // d0 56x56
    convL(wmma_conv_tile<4, 4>, Y, wp_d0, db0, X, 60, 56, 56, 64);
    hipMemsetAsync(Y, 0, bufBytes(68, 128), stream);                 // d1 60x60
    convL(wmma_conv_tile<4, 4>, X, wp_d1, db1, Y, 64, 60, 60, 68);

    // final: 60x60x128 -> 64x64x3, sigmoid, f32 NCHW
    wmma_deconv_final<<<(64 * 64 * 4) / 8, 256, 0, stream>>>(Y, wp_d2, db2, out);
}
